// RNNFamily_40810779247091
// MI455X (gfx1250) — matrix operand order repaired and checked by reading, not compiled, not run on hardware
//
#include <hip/hip_runtime.h>
#include <hip/hip_bf16.h>
#include <math.h>

// ---------------------------------------------------------------------------
// GRU (2 layers, B=64, T=512, D=256, H=512) for gfx1250 (MI455X, wave32, WMMA)
//
// Pipeline (all on `stream`, sequential):
//   1) convert X and the 4 weight matrices fp32 -> f16 in workspace
//   2) xw0 = f16( X_f16 @ W_ih0^T + b_ih0 )            [32768 x 1536] WMMA GEMM
//   3) layer-0 recurrence: 512 steps, 32 WGs; W_hh0 slice loaded into LDS once
//      via the Tensor Data Mover (tensor_load_to_lds + s_wait_tensorcnt);
//      grid barrier per step; writes h0_out f16 [32768 x 512]
//   4) xw1 = f16( h0_out @ W_ih1^T + b_ih1 )
//   5) layer-1 recurrence; at t == lengths[b]-1 writes final f32 h to d_out
// ---------------------------------------------------------------------------

typedef __attribute__((ext_vector_type(16))) _Float16     v16h;
typedef __attribute__((ext_vector_type(8)))  _Float16     v8h;
typedef __attribute__((ext_vector_type(8)))  float        v8f;
typedef __attribute__((ext_vector_type(4)))  unsigned int u32x4;
typedef __attribute__((ext_vector_type(8)))  int          i32x8;
typedef __attribute__((ext_vector_type(4)))  int          i32x4;

#define GRU_B 64
#define GRU_T 512
#define GRU_D 256
#define GRU_H 512
#define GRU_3H 1536

#if __has_include(<hip/amd_detail/amd_gfx1250_TDM.h>)
#define GRU_TDM_6ARG 1
#endif

// ---- WMMA helpers ----------------------------------------------------------

__device__ __forceinline__ v8f wmma_f16(v16h a, v16h b, v8f c) {
  // D = A(16x32 f16) x B(32x16 f16) + C(16x16 f32)
  return __builtin_amdgcn_wmma_f32_16x16x32_f16(
      /*neg_a=*/false, a, /*neg_b=*/false, b,
      /*c_mod=*/(short)0, c, /*reuse_a=*/false, /*reuse_b=*/false);
}

// A-fragment (16x32, f16): lanes 0-15 hold row M=lane with K = k..k+15;
// lanes 16-31 hold the same rows with K = k+16..k+31 (matches B fragments).
// Two contiguous 16B chunks.
__device__ __forceinline__ v16h frag_a(const _Float16* __restrict__ base,
                                       int m0, int ld, int k, int lane) {
  const int row = m0 + (lane & 15);
  const int kk  = k + ((lane >> 4) << 4);
  const _Float16* p = base + (size_t)row * ld + kk;
  v8h lo = *(const v8h*)(p);
  v8h hi = *(const v8h*)(p + 8);
  v16h r;
#pragma unroll
  for (int i = 0; i < 8; ++i) { r[i] = lo[i]; r[i + 8] = hi[i]; }
  return r;
}

// B-fragment (32x16, f16) for act @ W^T with W row-major [N,K]:
// lane holds column N = n0 + (lane&15); lanes 0-15 cover K = k..k+15,
// lanes 16-31 cover K = k+16..k+31 (16 contiguous halves = one 32B chunk).
__device__ __forceinline__ v16h frag_b_rowmajor(const _Float16* __restrict__ w,
                                                int n0, int ld, int k, int lane) {
  const int n  = n0 + (lane & 15);
  const int kk = k + ((lane >> 4) << 4);
  return *(const v16h*)(w + (size_t)n * ld + kk);
}

__device__ __forceinline__ float sigm(float x) { return 1.0f / (1.0f + __expf(-x)); }
__device__ __forceinline__ float tanh_fast(float x) { return 2.0f / (1.0f + __expf(-2.0f * x)) - 1.0f; }

// ---- Tensor Data Mover: 2D f16 tile Global -> LDS --------------------------
// Builds the D# descriptor per CDNA5 ISA and issues
// TENSOR_LOAD_TO_LDS. 2D tile: tile_d0 elements contiguous (dim0), tile_d1
// rows at stride0 elements. data_size = 2 bytes (f16).
__device__ __forceinline__ void tdm_load_2d_f16(unsigned lds_byte_off,
                                                const void* gptr,
                                                unsigned tile_d0,
                                                unsigned tile_d1,
                                                unsigned stride0_elems) {
  const unsigned long long ga = (unsigned long long)(uintptr_t)gptr;
  u32x4 g0;
  g0[0] = 1u;                                        // count=1, user descriptor
  g0[1] = lds_byte_off;                              // lds_addr (bytes)
  g0[2] = (unsigned)(ga & 0xFFFFFFFFu);              // global_addr[31:0]
  g0[3] = (unsigned)((ga >> 32) & 0x01FFFFFFu)       // global_addr[56:32]
        | (2u << 30);                                // type = 2 ("image")
  i32x8 g1;
  g1[0] = (int)(1u << 16);                           // data_size=1 (2B), mask=0
  g1[1] = (int)((tile_d0 & 0xFFFFu) << 16);          // tensor_dim0[15:0]
  g1[2] = (int)((tile_d0 >> 16) | ((tile_d1 & 0xFFFFu) << 16)); // td0hi|td1lo
  g1[3] = (int)((tile_d1 >> 16) | ((tile_d0 & 0xFFFFu) << 16)); // td1hi|tile_dim0
  g1[4] = (int)(tile_d1 & 0xFFFFu);                  // tile_dim1 | tile_dim2=0
  g1[5] = (int)stride0_elems;                        // tensor_dim0_stride[31:0]
  g1[6] = 0;                                         // stride0[47:32] | td1s lo
  g1[7] = 0;                                         // tensor_dim1_stride hi
  const i32x4 z4 = {0, 0, 0, 0};
#if defined(GRU_TDM_6ARG)
  const i32x8 z8 = {0, 0, 0, 0, 0, 0, 0, 0};
  __builtin_amdgcn_tensor_load_to_lds(g0, g1, z4, z4, z8, 0);
#else
  __builtin_amdgcn_tensor_load_to_lds(g0, g1, z4, z4, 0);
#endif
}

// ---- small utility kernels -------------------------------------------------

__global__ void cvt_f32_to_f16_kernel(const float* __restrict__ in,
                                      _Float16* __restrict__ out, int n) {
  int i = blockIdx.x * blockDim.x + threadIdx.x;
  if (i < n) out[i] = (_Float16)in[i];
}

__global__ void recur_init_kernel(_Float16* __restrict__ h, int n,
                                  unsigned* __restrict__ counter) {
  int i = blockIdx.x * blockDim.x + threadIdx.x;
  if (i < n) h[i] = (_Float16)0.0f;
  if (i == 0) *counter = 0u;
}

// ---- input-projection GEMM -------------------------------------------------
// out[M,N] (f16) = act[M,K] (f16) @ w[N,K]^T + bias[N] (f32), f32 accumulate.
// Block = 256 threads = 8 waves arranged 4(M) x 2(N); wave tile = 16M x 64N.
// K-loop is software pipelined: fragments for k+32 are fetched while the
// matrix pipe works on k, hiding global load latency behind v_wmma.
__global__ void __launch_bounds__(256)
gru_xw_gemm_kernel(const _Float16* __restrict__ act,
                   const _Float16* __restrict__ w,
                   const float* __restrict__ bias,
                   _Float16* __restrict__ out,
                   int M, int N, int K) {
  const int lane  = threadIdx.x & 31;
  const int wave  = threadIdx.x >> 5;
  const int m0    = blockIdx.x * 64 + (wave & 3) * 16;
  const int n0    = blockIdx.y * 128 + (wave >> 2) * 64;

  v8f acc[4] = {};
  v16h a = frag_a(act, m0, K, 0, lane);
  v16h b[4];
#pragma unroll
  for (int j = 0; j < 4; ++j) b[j] = frag_b_rowmajor(w, n0 + j * 16, K, 0, lane);

#pragma unroll 2
  for (int k = 0; k < K - 32; k += 32) {
    v16h an = frag_a(act, m0, K, k + 32, lane);
    v16h bn[4];
#pragma unroll
    for (int j = 0; j < 4; ++j)
      bn[j] = frag_b_rowmajor(w, n0 + j * 16, K, k + 32, lane);
#pragma unroll
    for (int j = 0; j < 4; ++j) acc[j] = wmma_f16(a, b[j], acc[j]);
    a = an;
#pragma unroll
    for (int j = 0; j < 4; ++j) b[j] = bn[j];
  }
#pragma unroll
  for (int j = 0; j < 4; ++j) acc[j] = wmma_f16(a, b[j], acc[j]);

  // C layout (7.12.2): VGPR i, lanes 0-15 -> (M=m0+i, N=n0+lane);
  // lanes 16-31 -> (M=m0+8+i, N=n0+lane-16).
  const int rbase = m0 + ((lane >> 4) << 3);
#pragma unroll
  for (int j = 0; j < 4; ++j) {
    const int col = n0 + j * 16 + (lane & 15);
    const float bv = bias[col];
#pragma unroll
    for (int i = 0; i < 8; ++i) {
      out[(size_t)(rbase + i) * N + col] = (_Float16)(acc[j][i] + bv);
    }
  }
}

// ---- recurrence ------------------------------------------------------------
// Grid = 32 WGs (each owns 16 hidden columns), block = 128 threads (4 waves,
// wave w owns batch rows 16w..16w+15). W_hh slice (48 rows x 512 f16 = 48 KB)
// is DMA'd into LDS once by the TDM and reused for all 512 steps.
// Grid-wide barrier per timestep via atomic counter.

__device__ __forceinline__ void grid_barrier(unsigned* counter, unsigned target) {
  __threadfence();
  __syncthreads();
  if (threadIdx.x == 0) {
    atomicAdd(counter, 1u);
    while (*(volatile unsigned*)counter < target) {
      __builtin_amdgcn_s_sleep(1);
    }
  }
  __syncthreads();
  __threadfence();
}

__global__ void __launch_bounds__(128)
gru_recur_kernel(const _Float16* __restrict__ xw,    // [B*T, 3H] f16 preacts
                 const _Float16* __restrict__ whh,   // [3H, H]   f16
                 const float*    __restrict__ bhh,   // [3H]      f32
                 _Float16* __restrict__ hbufA,       // [B, H] f16 (zeroed)
                 _Float16* __restrict__ hbufB,       // [B, H] f16
                 _Float16* __restrict__ hout,        // [B*T, H] f16 (layer out)
                 float*    __restrict__ final_out,   // [B, H] f32 (last layer)
                 const int* __restrict__ lengths,    // [B]
                 unsigned* __restrict__ barrier_cnt,
                 int store_all) {
  __shared__ alignas(32) _Float16 wl[48 * GRU_H];   // 48 KB

  const int lane = threadIdx.x & 31;
  const int wave = threadIdx.x >> 5;      // 0..3 -> batch tile
  const int c0   = blockIdx.x * 16;       // hidden-column slice
  const int m0   = wave * 16;
  const unsigned NWG = gridDim.x;

  // Stage the 3-gate W_hh slice for columns [c0, c0+16) into LDS via TDM:
  // one 16-row x 512-col f16 tile per gate (stride between rows = 512 elems).
  // Low 32 bits of the generic pointer == wave-relative LDS byte offset.
  if (wave == 0) {
    const unsigned lds_base = (unsigned)(uintptr_t)(void*)&wl[0];
#pragma unroll
    for (int g = 0; g < 3; ++g) {
      tdm_load_2d_f16(lds_base + (unsigned)(g * 16 * GRU_H) * 2u,
                      whh + (size_t)(g * GRU_H + c0) * GRU_H,
                      /*tile_d0=*/GRU_H, /*tile_d1=*/16,
                      /*stride0=*/GRU_H);
    }
    __builtin_amdgcn_s_wait_tensorcnt(0);
  }
  __syncthreads();

  const int col = c0 + (lane & 15);
  const float br = bhh[col];
  const float bz = bhh[GRU_H + col];
  const float bn = bhh[2 * GRU_H + col];
  const int rbase = m0 + ((lane >> 4) << 3);

  _Float16* hprev = hbufA;   // zero-initialized
  _Float16* hnext = hbufB;

  for (int t = 0; t < GRU_T; ++t) {
    v8f accr = {}, accz = {}, accn = {};
#pragma unroll
    for (int k = 0; k < GRU_H; k += 32) {
      v16h a = frag_a(hprev, m0, GRU_H, k, lane);
      // B fragments from LDS (ds_load_b128 pairs), rows 0/16/32 = r/z/n gates
      {
        const int kk = k + ((lane >> 4) << 4);
        v16h bR = *(const v16h*)(wl + (size_t)(0  + (lane & 15)) * GRU_H + kk);
        v16h bZ = *(const v16h*)(wl + (size_t)(16 + (lane & 15)) * GRU_H + kk);
        v16h bN = *(const v16h*)(wl + (size_t)(32 + (lane & 15)) * GRU_H + kk);
        accr = wmma_f16(a, bR, accr);
        accz = wmma_f16(a, bZ, accz);
        accn = wmma_f16(a, bN, accn);
      }
    }

#pragma unroll
    for (int i = 0; i < 8; ++i) {
      const int b = rbase + i;
      const _Float16* xwp = xw + (size_t)(b * GRU_T + t) * GRU_3H + col;
      if (t + 1 < GRU_T) __builtin_prefetch(xwp + GRU_3H, 0, 1);
      const float xr = (float)xwp[0];
      const float xz = (float)xwp[GRU_H];
      const float xn = (float)xwp[2 * GRU_H];
      const float r = sigm(xr + accr[i] + br);
      const float z = sigm(xz + accz[i] + bz);
      const float n = tanh_fast(xn + r * (accn[i] + bn));
      const float hp = (float)hprev[(size_t)b * GRU_H + col];
      const float hv = (1.0f - z) * n + z * hp;
      hnext[(size_t)b * GRU_H + col] = (_Float16)hv;
      if (store_all) {
        hout[(size_t)(b * GRU_T + t) * GRU_H + col] = (_Float16)hv;
      } else if (t == lengths[b] - 1) {
        final_out[(size_t)b * GRU_H + col] = hv;
      }
    }

    { _Float16* tmp = hprev; hprev = hnext; hnext = tmp; }
    grid_barrier(barrier_cnt, NWG * (unsigned)(t + 1));
  }
}

// ---------------------------------------------------------------------------

extern "C" void kernel_launch(void* const* d_in, const int* in_sizes, int n_in,
                              void* d_out, int out_size, void* d_ws, size_t ws_size,
                              hipStream_t stream) {
  (void)in_sizes; (void)n_in; (void)out_size; (void)ws_size;

  const float* X      = (const float*)d_in[0];
  const int*   lengths= (const int*)  d_in[1];
  const float* W_ih0  = (const float*)d_in[2];
  const float* W_hh0  = (const float*)d_in[3];
  const float* b_ih0  = (const float*)d_in[4];
  const float* b_hh0  = (const float*)d_in[5];
  const float* W_ih1  = (const float*)d_in[6];
  const float* W_hh1  = (const float*)d_in[7];
  const float* b_ih1  = (const float*)d_in[8];
  const float* b_hh1  = (const float*)d_in[9];
  float* out = (float*)d_out;

  // ---- workspace carve-up (256B aligned), total ~157 MB -------------------
  char* ws = (char*)d_ws;
  size_t off = 0;
  auto take = [&](size_t bytes) -> char* {
    char* p = ws + off;
    off = (off + bytes + 255) & ~(size_t)255;
    return p;
  };
  const int MT = GRU_B * GRU_T;                    // 32768 rows
  _Float16* x16    = (_Float16*)take((size_t)MT * GRU_D * 2);
  _Float16* wih0   = (_Float16*)take((size_t)GRU_3H * GRU_D * 2);
  _Float16* whh0   = (_Float16*)take((size_t)GRU_3H * GRU_H * 2);
  _Float16* wih1   = (_Float16*)take((size_t)GRU_3H * GRU_H * 2);
  _Float16* whh1   = (_Float16*)take((size_t)GRU_3H * GRU_H * 2);
  _Float16* xwbuf  = (_Float16*)take((size_t)MT * GRU_3H * 2);
  _Float16* h0out  = (_Float16*)take((size_t)MT * GRU_H * 2);
  _Float16* hA     = (_Float16*)take((size_t)GRU_B * GRU_H * 2);
  _Float16* hB     = (_Float16*)take((size_t)GRU_B * GRU_H * 2);
  unsigned* barcnt = (unsigned*)take(256);

  auto cvt = [&](const float* src, _Float16* dst, int n) {
    cvt_f32_to_f16_kernel<<<(n + 255) / 256, 256, 0, stream>>>(src, dst, n);
  };

  // 1) fp32 -> f16 conversions
  cvt(X,     x16,  MT * GRU_D);
  cvt(W_ih0, wih0, GRU_3H * GRU_D);
  cvt(W_hh0, whh0, GRU_3H * GRU_H);
  cvt(W_ih1, wih1, GRU_3H * GRU_H);
  cvt(W_hh1, whh1, GRU_3H * GRU_H);

  // 2) xw0 = X @ W_ih0^T + b_ih0
  {
    dim3 grid(MT / 64, GRU_3H / 128);
    gru_xw_gemm_kernel<<<grid, 256, 0, stream>>>(x16, wih0, b_ih0, xwbuf,
                                                 MT, GRU_3H, GRU_D);
  }

  // 3) layer-0 recurrence (writes all hidden states)
  recur_init_kernel<<<(GRU_B * GRU_H + 255) / 256, 256, 0, stream>>>(
      hA, GRU_B * GRU_H, barcnt);
  gru_recur_kernel<<<GRU_H / 16, 128, 0, stream>>>(
      xwbuf, whh0, b_hh0, hA, hB, h0out, nullptr, lengths, barcnt,
      /*store_all=*/1);

  // 4) xw1 = h0 @ W_ih1^T + b_ih1
  {
    dim3 grid(MT / 64, GRU_3H / 128);
    gru_xw_gemm_kernel<<<grid, 256, 0, stream>>>(h0out, wih1, b_ih1, xwbuf,
                                                 MT, GRU_3H, GRU_H);
  }

  // 5) layer-1 recurrence (emits h at t = lengths[b]-1 into d_out, f32)
  recur_init_kernel<<<(GRU_B * GRU_H + 255) / 256, 256, 0, stream>>>(
      hA, GRU_B * GRU_H, barcnt);
  gru_recur_kernel<<<GRU_H / 16, 128, 0, stream>>>(
      xwbuf, whh1, b_hh1, hA, hB, nullptr, out, lengths, barcnt,
      /*store_all=*/0);
}
